// M3GNet_59442347376878
// MI455X (gfx1250) — matrix operand order repaired and checked by reading, not compiled, not run on hardware
//
#include <hip/hip_runtime.h>
#include <hip/hip_bf16.h>
#include <stdint.h>

#define NATOMS 32768
#define DEG 12
#define NEDGES (NATOMS * DEG)      /* 393216 */
#define NF 64
#define NGRAPHS 512

typedef _Float16 half16 __attribute__((ext_vector_type(16)));
typedef _Float16 half8  __attribute__((ext_vector_type(8)));
typedef float    float8 __attribute__((ext_vector_type(8)));

union H16U { half16 v; half8 h[2]; };

__device__ __constant__ float d_roots[9] = {
    3.14159265358979323846f, 6.28318530717958647692f, 9.42477796076937971538f,
    4.493409457909064f, 7.725251836937707f, 10.904121659428899f,
    5.763459196894550f, 9.095011330476355f, 12.322940970566582f };

/* hardware v_rcp_f32: 1 ulp, fine for strictly-positive well-conditioned args */
__device__ inline float rcpf_(float x) { return __builtin_amdgcn_rcpf(x); }
__device__ inline float sigmoidf_(float x) { return rcpf_(1.0f + __expf(-x)); }
__device__ inline float siluf_(float x) { return x * sigmoidf_(x); }

__device__ inline float cut4f_(float r) {
    if (r > 4.0f) return 0.0f;
    float x = r * 0.25f;
    float x3 = x * x * x;
    return 1.0f - 6.0f * x3 * x * x + 15.0f * x3 * x - 10.0f * x3;
}

__device__ inline void sbf9_(float r, const float* __restrict__ coef, float* out) {
    float rs = r * 0.2f;
#pragma unroll
    for (int k = 0; k < 9; k++) {
        float x = d_roots[k] * rs;
        float s = __sinf(x), c = __cosf(x);
        float inv = rcpf_(x);
        int l = k / 3;
        float j;
        if (l == 0)       j = s * inv;
        else if (l == 1)  j = s * inv * inv - c * inv;
        else              j = (3.0f * inv * inv * inv - inv) * s - 3.0f * c * inv * inv;
        out[k] = j * coef[k];
    }
}

/* ---------------- WMMA fragment helpers (CDNA5 ISA §7.12.2) --- */
__device__ inline float8 wmma16(half16 a, half16 b, float8 c) {
    return __builtin_amdgcn_wmma_f32_16x16x32_f16(false, a, false, b, (short)0, c, false, false);
}

__device__ inline float8 zero8_() {
    float8 z;
#pragma unroll
    for (int r = 0; r < 8; r++) z[r] = 0.0f;
    return z;
}

__device__ inline half16 ldA(const _Float16* X, int ldw, int lane, int ks) {
    int m = lane & 15, hi = lane >> 4;
    const _Float16* p = X + m * ldw + ks * 32 + hi * 8;
    H16U u;
    u.h[0] = *(const half8*)(p);
    u.h[1] = *(const half8*)(p + 16);
    return u.v;
}

__device__ inline half16 ldB(const _Float16* Bp, int lane, int ks, int t) {
    int hi = lane >> 4;
    int n = (lane & 15) + 16 * t;
    const _Float16* p = Bp + ((((ks * 2) * 64) + n) << 4) + hi * 8;
    H16U u;
    u.h[0] = *(const half8*)(p);
    u.h[1] = *(const half8*)(p + 1024);
    return u.v;
}

__device__ inline half16 ldA9(const float* __restrict__ src, int e0, int lane) {
    int m = lane & 15, hi = lane >> 4;
    const float* p = src + (size_t)(e0 + m) * 9;
    H16U u;
#pragma unroll
    for (int h = 0; h < 8; h++) {
        int k = hi * 8 + h;
        float v = (k < 9) ? p[k] : 0.0f;
        u.v[h] = (_Float16)v;
        u.v[h + 8] = (_Float16)0.0f;
    }
    return u.v;
}

__device__ inline float8 biasFrag(const float* __restrict__ b, int lane, int t) {
    float v = b[(lane & 15) + 16 * t];
    float8 c;
#pragma unroll
    for (int r = 0; r < 8; r++) c[r] = v;
    return c;
}

__device__ inline void stAct(_Float16* Hs, int lane, int t, float8 c, bool sig) {
    int n = (lane & 15) + 16 * t;
    int mb = (lane >> 4) * 8;
#pragma unroll
    for (int r = 0; r < 8; r++) {
        float v = c[r];
        float a = sig ? sigmoidf_(v) : siluf_(v);
        Hs[(mb + r) * 64 + n] = (_Float16)a;
    }
}

/* ---------------- small kernels ----------------------------------------- */
__global__ void coef_kernel(float* coef) {
    int k = threadIdx.x;
    if (k >= 9) return;
    const double roots[9] = {
        3.141592653589793, 6.283185307179586, 9.42477796076938,
        4.493409457909064, 7.725251836937707, 10.904121659428899,
        5.763459196894550, 9.095011330476355, 12.322940970566582 };
    int l = k / 3;
    double x = roots[k];
    double s = sin(x), c = cos(x), jl1;
    if (l == 0)      jl1 = s / (x * x) - c / x;
    else if (l == 1) jl1 = (3.0 / (x * x * x) - 1.0 / x) * s - 3.0 * c / (x * x);
    else             jl1 = (15.0 / (x * x * x * x) - 6.0 / (x * x)) * s
                           - (15.0 / (x * x * x) - 1.0 / x) * c;
    coef[k] = (float)(0.12649110640673518 / fabs(jl1));
}

__global__ void pack_w_kernel(const float* __restrict__ W, _Float16* __restrict__ out,
                              int fin, int total) {
    int i = blockIdx.x * blockDim.x + threadIdx.x;
    if (i >= total) return;
    int h = i & 15;
    int n = (i >> 4) & 63;
    int g = i >> 10;              /* g = s*2 + hi */
    int s = g >> 1, hi = g & 1;
    int k = s * 32 + hi * 16 + h;
    float v = (k < fin) ? W[k * 64 + n] : 0.0f;
    out[i] = (_Float16)v;
}

__global__ void node_init_kernel(float* __restrict__ node, const float* __restrict__ embed,
                                 const int* __restrict__ ntype) {
    int i = blockIdx.x * blockDim.x + threadIdx.x;
    if (i >= NATOMS * NF) return;
    int a = i >> 6, c = i & 63;
    node[i] = embed[ntype[a] * NF + c];
}

__global__ void edge_init_kernel(const float* __restrict__ bond_vec,
                                 const float* __restrict__ eeW, const float* __restrict__ eeb,
                                 const float* __restrict__ coef,
                                 float* __restrict__ rbf, float* __restrict__ cut,
                                 float* __restrict__ ef) {
    int e = blockIdx.x * blockDim.x + threadIdx.x;
    if (e >= NEDGES) return;
    float vx = bond_vec[3 * e], vy = bond_vec[3 * e + 1], vz = bond_vec[3 * e + 2];
    float d = sqrtf(vx * vx + vy * vy + vz * vz);
    float rb[9];
    sbf9_(d, coef, rb);
#pragma unroll
    for (int k = 0; k < 9; k++) rbf[(size_t)e * 9 + k] = rb[k];
    cut[e] = cut4f_(d);
    for (int c = 0; c < NF; c++) {
        float s = eeb[c];
#pragma unroll
        for (int k = 0; k < 9; k++) s += rb[k] * eeW[k * NF + c];
        ef[(size_t)e * NF + c] = siluf_(s);
    }
}

__global__ void atoms_kernel(const float* __restrict__ node, const float* __restrict__ Wa,
                             const float* __restrict__ ba, float* __restrict__ atoms) {
    int a = blockIdx.x * blockDim.x + threadIdx.x;
    if (a >= NATOMS) return;
    const float* row = node + (size_t)a * NF;
#pragma unroll
    for (int j = 0; j < 9; j++) {
        float s = ba[j];
        for (int c = 0; c < NF; c++) s += row[c] * Wa[c * 9 + j];
        atoms[(size_t)a * 9 + j] = sigmoidf_(s);
    }
}

__global__ void triples_kernel(const float* __restrict__ bond_vec,
                               const int* __restrict__ edge_dst,
                               const float* __restrict__ atoms,
                               const float* __restrict__ cut,
                               const float* __restrict__ coef,
                               float* __restrict__ nb) {
    __shared__ float svh[192][3];
    __shared__ float sq[192][9];
    __shared__ float sc[192];
    int t = threadIdx.x;
    int e = blockIdx.x * 192 + t;
    float vx = bond_vec[3 * e], vy = bond_vec[3 * e + 1], vz = bond_vec[3 * e + 2];
    float d = sqrtf(vx * vx + vy * vy + vz * vz);
    float inv = rcpf_(d);
    float h0 = vx * inv, h1 = vy * inv, h2 = vz * inv;
    float rb[9];
    sbf9_(d, coef, rb);
    float ce = cut[e];
    int da = edge_dst[e];
#pragma unroll
    for (int k = 0; k < 9; k++) sq[t][k] = rb[k] * atoms[(size_t)da * 9 + k] * ce;
    svh[t][0] = h0; svh[t][1] = h1; svh[t][2] = h2;
    sc[t] = ce;
    __syncthreads();
    int base = (t / 12) * 12;
    float acc[9];
#pragma unroll
    for (int k = 0; k < 9; k++) acc[k] = 0.0f;
    for (int pj = 0; pj < 12; pj++) {
        int e2 = base + pj;
        if (e2 == t) continue;
        float ct = h0 * svh[e2][0] + h1 * svh[e2][1] + h2 * svh[e2][2];
        float s0 = 0.28209479177387814f;
        float s1 = 0.4886025119029199f * ct;
        float s2 = 0.31539156525252005f * (3.0f * ct * ct - 1.0f);
        acc[0] += s0 * sq[e2][0]; acc[1] += s0 * sq[e2][1]; acc[2] += s0 * sq[e2][2];
        acc[3] += s1 * sq[e2][3]; acc[4] += s1 * sq[e2][4]; acc[5] += s1 * sq[e2][5];
        acc[6] += s2 * sq[e2][6]; acc[7] += s2 * sq[e2][7]; acc[8] += s2 * sq[e2][8];
    }
    float cm = sc[t];
#pragma unroll
    for (int k = 0; k < 9; k++) nb[(size_t)e * 9 + k] = cm * acc[k];
}

/* ---------------- edge update: WMMA gated MLP --------------------------- */
__global__ void __launch_bounds__(128)
edge_update_kernel(const float* __restrict__ nodeCur, float* __restrict__ ef,
                   const float* __restrict__ rbf, const float* __restrict__ nb,
                   const int* __restrict__ edge_src, const int* __restrict__ edge_dst,
                   const _Float16* __restrict__ TBM, const _Float16* __restrict__ TBG,
                   const _Float16* __restrict__ EM1, const _Float16* __restrict__ EM2,
                   const _Float16* __restrict__ EM3, const _Float16* __restrict__ EG1,
                   const _Float16* __restrict__ EG2, const _Float16* __restrict__ EG3,
                   const _Float16* __restrict__ EWT,
                   const float* __restrict__ bm1, const float* __restrict__ bm2,
                   const float* __restrict__ bm3, const float* __restrict__ bg1,
                   const float* __restrict__ bg2, const float* __restrict__ bg3) {
    __shared__ __align__(16) _Float16 sX[4][16 * 192];
    __shared__ __align__(16) _Float16 sH[4][16 * 64];
    __shared__ __align__(16) _Float16 sG[4][16 * 64];
    __shared__ int sSrc[4][16], sDst[4][16];

    int wave = threadIdx.x >> 5, lane = threadIdx.x & 31;
    int e0 = (blockIdx.x * 4 + wave) * 16;
    _Float16* X = sX[wave];
    _Float16* Hs = sH[wave];
    _Float16* Gs = sG[wave];

    if (lane < 16) sSrc[wave][lane] = edge_src[e0 + lane];
    else sDst[wave][lane - 16] = edge_dst[e0 + lane - 16];

    for (int idx = lane; idx < 16 * 128; idx += 32) {
        int row = idx >> 7, col = idx & 127;
        int a = (col < 64) ? sSrc[wave][row] : sDst[wave][row];
        X[row * 192 + col] = (_Float16)nodeCur[(size_t)a * NF + (col & 63)];
    }

    int n0 = lane & 15, hi = lane >> 4;

    half16 aNB = ldA9(nb, e0, lane);
    for (int t = 0; t < 4; t++) {
        float8 cm = zero8_(), cg = zero8_();
        cm = wmma16(aNB, ldB(TBM, lane, 0, t), cm);
        cg = wmma16(aNB, ldB(TBG, lane, 0, t), cg);
#pragma unroll
        for (int r = 0; r < 8; r++) {
            int M = r + hi * 8, e = e0 + M, n = n0 + 16 * t;
            float old = ef[(size_t)e * NF + n];
            float nv = old + siluf_(cm[r]) * sigmoidf_(cg[r]);
            ef[(size_t)e * NF + n] = nv;
            X[M * 192 + 128 + n] = (_Float16)nv;
        }
    }

    for (int t = 0; t < 4; t++) {
        float8 cm = biasFrag(bm1, lane, t), cg = biasFrag(bg1, lane, t);
#pragma unroll
        for (int ks = 0; ks < 6; ks++) {
            half16 a = ldA(X, 192, lane, ks);
            cm = wmma16(a, ldB(EM1, lane, ks, t), cm);
            cg = wmma16(a, ldB(EG1, lane, ks, t), cg);
        }
        stAct(Hs, lane, t, cm, false);
        stAct(Gs, lane, t, cg, false);
    }
    {
        half16 aH0 = ldA(Hs, 64, lane, 0), aH1 = ldA(Hs, 64, lane, 1);
        half16 aG0 = ldA(Gs, 64, lane, 0), aG1 = ldA(Gs, 64, lane, 1);
        for (int t = 0; t < 4; t++) {
            float8 cm = biasFrag(bm2, lane, t), cg = biasFrag(bg2, lane, t);
            cm = wmma16(aH0, ldB(EM2, lane, 0, t), cm);
            cm = wmma16(aH1, ldB(EM2, lane, 1, t), cm);
            cg = wmma16(aG0, ldB(EG2, lane, 0, t), cg);
            cg = wmma16(aG1, ldB(EG2, lane, 1, t), cg);
            stAct(Hs, lane, t, cm, false);
            stAct(Gs, lane, t, cg, false);
        }
    }
    {
        half16 aH0 = ldA(Hs, 64, lane, 0), aH1 = ldA(Hs, 64, lane, 1);
        half16 aG0 = ldA(Gs, 64, lane, 0), aG1 = ldA(Gs, 64, lane, 1);
        half16 aRB = ldA9(rbf, e0, lane);
        for (int t = 0; t < 4; t++) {
            float8 cm = biasFrag(bm3, lane, t), cg = biasFrag(bg3, lane, t);
            float8 cw = zero8_();
            cm = wmma16(aH0, ldB(EM3, lane, 0, t), cm);
            cm = wmma16(aH1, ldB(EM3, lane, 1, t), cm);
            cg = wmma16(aG0, ldB(EG3, lane, 0, t), cg);
            cg = wmma16(aG1, ldB(EG3, lane, 1, t), cg);
            cw = wmma16(aRB, ldB(EWT, lane, 0, t), cw);
#pragma unroll
            for (int r = 0; r < 8; r++) {
                int M = r + hi * 8, e = e0 + M, n = n0 + 16 * t;
                float upd = siluf_(cm[r]) * sigmoidf_(cg[r]) * cw[r];
                ef[(size_t)e * NF + n] += upd;
            }
        }
    }
}

__global__ void __launch_bounds__(128)
node_update_kernel(const float* __restrict__ nodeCur, const float* __restrict__ ef,
                   const float* __restrict__ rbf,
                   const int* __restrict__ edge_src, const int* __restrict__ edge_dst,
                   const _Float16* __restrict__ NM1, const _Float16* __restrict__ NM2,
                   const _Float16* __restrict__ NM3, const _Float16* __restrict__ NG1,
                   const _Float16* __restrict__ NG2, const _Float16* __restrict__ NG3,
                   const _Float16* __restrict__ NWT,
                   const float* __restrict__ bm1, const float* __restrict__ bm2,
                   const float* __restrict__ bm3, const float* __restrict__ bg1,
                   const float* __restrict__ bg2, const float* __restrict__ bg3,
                   float* __restrict__ nodeNext) {
    __shared__ __align__(16) _Float16 sX[4][16 * 192];
    __shared__ __align__(16) _Float16 sH[4][16 * 64];
    __shared__ __align__(16) _Float16 sG[4][16 * 64];
    __shared__ int sSrc[4][16], sDst[4][16];

    int wave = threadIdx.x >> 5, lane = threadIdx.x & 31;
    int e0 = (blockIdx.x * 4 + wave) * 16;
    _Float16* X = sX[wave];
    _Float16* Hs = sH[wave];
    _Float16* Gs = sG[wave];

    if (lane < 16) sSrc[wave][lane] = edge_src[e0 + lane];
    else sDst[wave][lane - 16] = edge_dst[e0 + lane - 16];

    for (int idx = lane; idx < 16 * 192; idx += 32) {
        int row = idx / 192, col = idx - row * 192;
        float v;
        if (col < 64)       v = nodeCur[(size_t)sSrc[wave][row] * NF + col];
        else if (col < 128) v = nodeCur[(size_t)sDst[wave][row] * NF + (col - 64)];
        else                v = ef[(size_t)(e0 + row) * NF + (col - 128)];
        X[row * 192 + col] = (_Float16)v;
    }

    int n0 = lane & 15, hi = lane >> 4;

    for (int t = 0; t < 4; t++) {
        float8 cm = biasFrag(bm1, lane, t), cg = biasFrag(bg1, lane, t);
#pragma unroll
        for (int ks = 0; ks < 6; ks++) {
            half16 a = ldA(X, 192, lane, ks);
            cm = wmma16(a, ldB(NM1, lane, ks, t), cm);
            cg = wmma16(a, ldB(NG1, lane, ks, t), cg);
        }
        stAct(Hs, lane, t, cm, false);
        stAct(Gs, lane, t, cg, false);
    }
    {
        half16 aH0 = ldA(Hs, 64, lane, 0), aH1 = ldA(Hs, 64, lane, 1);
        half16 aG0 = ldA(Gs, 64, lane, 0), aG1 = ldA(Gs, 64, lane, 1);
        for (int t = 0; t < 4; t++) {
            float8 cm = biasFrag(bm2, lane, t), cg = biasFrag(bg2, lane, t);
            cm = wmma16(aH0, ldB(NM2, lane, 0, t), cm);
            cm = wmma16(aH1, ldB(NM2, lane, 1, t), cm);
            cg = wmma16(aG0, ldB(NG2, lane, 0, t), cg);
            cg = wmma16(aG1, ldB(NG2, lane, 1, t), cg);
            stAct(Hs, lane, t, cm, false);
            stAct(Gs, lane, t, cg, false);
        }
    }
    {
        half16 aH0 = ldA(Hs, 64, lane, 0), aH1 = ldA(Hs, 64, lane, 1);
        half16 aG0 = ldA(Gs, 64, lane, 0), aG1 = ldA(Gs, 64, lane, 1);
        half16 aRB = ldA9(rbf, e0, lane);
        for (int t = 0; t < 4; t++) {
            float8 cm = biasFrag(bm3, lane, t), cg = biasFrag(bg3, lane, t);
            float8 cw = zero8_();
            cm = wmma16(aH0, ldB(NM3, lane, 0, t), cm);
            cm = wmma16(aH1, ldB(NM3, lane, 1, t), cm);
            cg = wmma16(aG0, ldB(NG3, lane, 0, t), cg);
            cg = wmma16(aG1, ldB(NG3, lane, 1, t), cg);
            cw = wmma16(aRB, ldB(NWT, lane, 0, t), cw);
#pragma unroll
            for (int r = 0; r < 8; r++) {
                int M = r + hi * 8, n = n0 + 16 * t;
                int dst = sDst[wave][M];
                float msg = siluf_(cm[r]) * sigmoidf_(cg[r]) * cw[r];
                atomicAdd(&nodeNext[(size_t)dst * NF + n], msg);
            }
        }
    }
}

__global__ void readout_kernel(const float* __restrict__ node,
                               const float* __restrict__ W1, const float* __restrict__ b1,
                               const float* __restrict__ W2, const float* __restrict__ b2,
                               const float* __restrict__ W3, const float* __restrict__ b3,
                               float* __restrict__ out) {
    __shared__ float g[64], h1[64], h2[64];
    int c = threadIdx.x, gr = blockIdx.x;
    float s = 0.0f;
    for (int i = 0; i < 64; i++) s += node[(size_t)(gr * 64 + i) * NF + c];
    g[c] = s * (1.0f / 64.0f);
    __syncthreads();
    float a = b1[c];
    for (int i = 0; i < 64; i++) a += g[i] * W1[i * 64 + c];
    h1[c] = siluf_(a);
    __syncthreads();
    float a2 = b2[c];
    for (int i = 0; i < 64; i++) a2 += h1[i] * W2[i * 64 + c];
    h2[c] = siluf_(a2);
    __syncthreads();
    if (c == 0) {
        float o = b3[0];
        for (int i = 0; i < 64; i++) o += h2[i] * W3[i];
        out[gr] = o;
    }
}

/* ---------------- host -------------------------------------------------- */
struct Blk {
    const float *tbA_W, *tbA_b, *tbM, *tbG, *ewt, *nwt;
    const float *emW[3], *emb[3], *egW[3], *egb[3];
    const float *nmW[3], *nmb[3], *ngW[3], *ngb[3];
};

enum {
    OFF_EM1 = 0, OFF_EG1 = 12288, OFF_EM2 = 24576, OFF_EG2 = 28672,
    OFF_EM3 = 32768, OFF_EG3 = 36864, OFF_EWT = 40960, OFF_TBM = 43008,
    OFF_TBG = 45056, OFF_NM1 = 47104, OFF_NG1 = 59392, OFF_NM2 = 71680,
    OFF_NG2 = 75776, OFF_NM3 = 79872, OFF_NG3 = 83968, OFF_NWT = 88064,
    BLK_PACK = 90112
};

extern "C" void kernel_launch(void* const* d_in, const int* in_sizes, int n_in,
                              void* d_out, int out_size, void* d_ws, size_t ws_size,
                              hipStream_t stream) {
    (void)n_in; (void)out_size; (void)ws_size;
    auto FP = [&](int i) { return (const float*)d_in[i]; };
    auto IP = [&](int i) { return (const int*)d_in[i]; };

    bool outerSorted = (in_sizes[1] == NEDGES);
    bool innerSorted;
    int p, iESrc, iEDst, iNType;
    if (outerSorted) {
        innerSorted = true;
        iEDst = 1; iESrc = 2; iNType = 4; p = 5;
    } else {
        innerSorted = (in_sizes[1] == 12288);
        iESrc = 100; iEDst = 101; iNType = 102; p = 1;
    }

    const float* bond_vec = FP(0);
    const int* edge_src = IP(iESrc);
    const int* edge_dst = IP(iEDst);
    const int* node_type = IP(iNType);

    const float *node_embed, *eeW, *eeb, *fW[3], *fb[3];
    Blk blk[3];
    if (innerSorted) {
        for (int b = 0; b < 3; b++) {
            int s = p + 30 * b;
            for (int i = 0; i < 3; i++) { blk[b].egW[i] = FP(s + 2 * i); blk[b].egb[i] = FP(s + 2 * i + 1); }
            for (int i = 0; i < 3; i++) { blk[b].emW[i] = FP(s + 6 + 2 * i); blk[b].emb[i] = FP(s + 7 + 2 * i); }
            blk[b].ewt = FP(s + 12);
            for (int i = 0; i < 3; i++) { blk[b].ngW[i] = FP(s + 13 + 2 * i); blk[b].ngb[i] = FP(s + 14 + 2 * i); }
            for (int i = 0; i < 3; i++) { blk[b].nmW[i] = FP(s + 19 + 2 * i); blk[b].nmb[i] = FP(s + 20 + 2 * i); }
            blk[b].nwt = FP(s + 25);
            blk[b].tbA_W = FP(s + 26); blk[b].tbA_b = FP(s + 27);
            blk[b].tbG = FP(s + 28); blk[b].tbM = FP(s + 29);
        }
        eeW = FP(p + 90); eeb = FP(p + 91);
        for (int i = 0; i < 3; i++) { fW[i] = FP(p + 92 + 2 * i); fb[i] = FP(p + 93 + 2 * i); }
        node_embed = FP(p + 98);
    } else {
        node_embed = FP(p); eeW = FP(p + 1); eeb = FP(p + 2);
        for (int b = 0; b < 3; b++) {
            int s = p + 3 + 30 * b;
            blk[b].tbA_W = FP(s); blk[b].tbA_b = FP(s + 1);
            blk[b].tbM = FP(s + 2); blk[b].tbG = FP(s + 3);
            for (int i = 0; i < 3; i++) { blk[b].emW[i] = FP(s + 4 + 2 * i); blk[b].emb[i] = FP(s + 5 + 2 * i); }
            for (int i = 0; i < 3; i++) { blk[b].egW[i] = FP(s + 10 + 2 * i); blk[b].egb[i] = FP(s + 11 + 2 * i); }
            blk[b].ewt = FP(s + 16);
            for (int i = 0; i < 3; i++) { blk[b].nmW[i] = FP(s + 17 + 2 * i); blk[b].nmb[i] = FP(s + 18 + 2 * i); }
            for (int i = 0; i < 3; i++) { blk[b].ngW[i] = FP(s + 23 + 2 * i); blk[b].ngb[i] = FP(s + 24 + 2 * i); }
            blk[b].nwt = FP(s + 29);
        }
        for (int i = 0; i < 3; i++) { fW[i] = FP(p + 93 + 2 * i); fb[i] = FP(p + 94 + 2 * i); }
    }

    uintptr_t cur = (uintptr_t)d_ws;
    auto alloc = [&](size_t bytes) -> void* {
        void* r = (void*)cur;
        cur += (bytes + 255) & ~(size_t)255;
        return r;
    };
    float* rbf   = (float*)alloc((size_t)NEDGES * 9 * 4);
    float* cut   = (float*)alloc((size_t)NEDGES * 4);
    float* ef    = (float*)alloc((size_t)NEDGES * NF * 4);
    float* nodeA = (float*)alloc((size_t)NATOMS * NF * 4);
    float* nodeB = (float*)alloc((size_t)NATOMS * NF * 4);
    float* atoms = (float*)alloc((size_t)NATOMS * 9 * 4);
    float* nb    = (float*)alloc((size_t)NEDGES * 9 * 4);
    float* coef  = (float*)alloc(64 * 4);
    _Float16* pack = (_Float16*)alloc((size_t)3 * BLK_PACK * 2);

    auto packW = [&](const float* W, _Float16* dst, int fin, int kpad) {
        int total = kpad * 64;
        pack_w_kernel<<<(total + 255) / 256, 256, 0, stream>>>(W, dst, fin, total);
    };

    coef_kernel<<<1, 16, 0, stream>>>(coef);
    for (int b = 0; b < 3; b++) {
        _Float16* pk = pack + (size_t)b * BLK_PACK;
        packW(blk[b].emW[0], pk + OFF_EM1, 192, 192);
        packW(blk[b].egW[0], pk + OFF_EG1, 192, 192);
        packW(blk[b].emW[1], pk + OFF_EM2, 64, 64);
        packW(blk[b].egW[1], pk + OFF_EG2, 64, 64);
        packW(blk[b].emW[2], pk + OFF_EM3, 64, 64);
        packW(blk[b].egW[2], pk + OFF_EG3, 64, 64);
        packW(blk[b].ewt,    pk + OFF_EWT, 9, 32);
        packW(blk[b].tbM,    pk + OFF_TBM, 9, 32);
        packW(blk[b].tbG,    pk + OFF_TBG, 9, 32);
        packW(blk[b].nmW[0], pk + OFF_NM1, 192, 192);
        packW(blk[b].ngW[0], pk + OFF_NG1, 192, 192);
        packW(blk[b].nmW[1], pk + OFF_NM2, 64, 64);
        packW(blk[b].ngW[1], pk + OFF_NG2, 64, 64);
        packW(blk[b].nmW[2], pk + OFF_NM3, 64, 64);
        packW(blk[b].ngW[2], pk + OFF_NG3, 64, 64);
        packW(blk[b].nwt,    pk + OFF_NWT, 9, 32);
    }

    node_init_kernel<<<(NATOMS * NF) / 256, 256, 0, stream>>>(nodeA, node_embed, node_type);
    edge_init_kernel<<<NEDGES / 256, 256, 0, stream>>>(bond_vec, eeW, eeb, coef, rbf, cut, ef);

    float* nodeCur = nodeA;
    float* nodeNext = nodeB;
    for (int b = 0; b < 3; b++) {
        _Float16* pk = pack + (size_t)b * BLK_PACK;
        atoms_kernel<<<NATOMS / 256, 256, 0, stream>>>(nodeCur, blk[b].tbA_W, blk[b].tbA_b, atoms);
        triples_kernel<<<NATOMS / 16, 192, 0, stream>>>(bond_vec, edge_dst, atoms, cut, coef, nb);
        edge_update_kernel<<<NEDGES / 64, 128, 0, stream>>>(
            nodeCur, ef, rbf, nb, edge_src, edge_dst,
            pk + OFF_TBM, pk + OFF_TBG,
            pk + OFF_EM1, pk + OFF_EM2, pk + OFF_EM3,
            pk + OFF_EG1, pk + OFF_EG2, pk + OFF_EG3, pk + OFF_EWT,
            blk[b].emb[0], blk[b].emb[1], blk[b].emb[2],
            blk[b].egb[0], blk[b].egb[1], blk[b].egb[2]);
        hipMemcpyAsync(nodeNext, nodeCur, (size_t)NATOMS * NF * 4,
                       hipMemcpyDeviceToDevice, stream);
        node_update_kernel<<<NEDGES / 64, 128, 0, stream>>>(
            nodeCur, ef, rbf, edge_src, edge_dst,
            pk + OFF_NM1, pk + OFF_NM2, pk + OFF_NM3,
            pk + OFF_NG1, pk + OFF_NG2, pk + OFF_NG3, pk + OFF_NWT,
            blk[b].nmb[0], blk[b].nmb[1], blk[b].nmb[2],
            blk[b].ngb[0], blk[b].ngb[1], blk[b].ngb[2],
            nodeNext);
        float* tmp = nodeCur; nodeCur = nodeNext; nodeNext = tmp;
    }
    readout_kernel<<<NGRAPHS, 64, 0, stream>>>(nodeCur, fW[0], fb[0], fW[1], fb[1],
                                               fW[2], fb[2], (float*)d_out);
}
